// RetinaFace_BBT_12051678233267
// MI455X (gfx1250) — matrix operand order repaired and checked by reading, not compiled, not run on hardware
//
#include <hip/hip_runtime.h>
#include <hip/hip_bf16.h>
#include <stdint.h>

typedef _Float16 half_t;
typedef __attribute__((ext_vector_type(16))) _Float16 v16h;
typedef __attribute__((ext_vector_type(8)))  _Float16 v8h;
typedef __attribute__((ext_vector_type(8)))  float    v8f;

#define BATCH 4
#define ATOT  102300
#define NSORT 131072
#define KTOP  512

// ---------------- zero fill (f16, 8-wide) ------------------------------------
__global__ void rf_zero_f16(half_t* __restrict__ p, long n8) {  // n8 = n/8
  long i = (long)blockIdx.x * blockDim.x + threadIdx.x;
  if (i < n8) *(v8h*)(p + i * 8) = (v8h){};
}

// ---------------- weight packing: OIHW f32 -> [tap][chunk][coutP][32] f16 -----
__global__ void rf_pack_w(const float* __restrict__ w, half_t* __restrict__ wp,
                          int Cin, int Cout, int CoutP, int taps) {
  int chunks = Cin >> 5;
  long total = (long)taps * chunks * CoutP * 32;
  long i = (long)blockIdx.x * blockDim.x + threadIdx.x;
  if (i >= total) return;
  int k = (int)(i & 31);
  long r = i >> 5;
  int n = (int)(r % CoutP); r /= CoutP;
  int c = (int)(r % chunks);
  int tap = (int)(r / chunks);
  float v = 0.f;
  if (n < Cout) {
    int ci = c * 32 + k;
    long off = (taps == 9) ? (((long)n * Cin + ci) * 9 + tap) : ((long)n * Cin + ci);
    v = w[off];
  }
  wp[i] = (half_t)v;
}

__global__ void rf_pack_b(const float* __restrict__ b, float* __restrict__ bp,
                          int Cout, int CoutP) {
  int i = blockIdx.x * blockDim.x + threadIdx.x;
  if (i < CoutP) bp[i] = (i < Cout) ? b[i] : 0.f;
}

// ---------------- NCHW f32 -> halo-padded NHWC f16 ---------------------------
__global__ void rf_nchw_to_nhwc_pad(const float* __restrict__ x, half_t* __restrict__ xh,
                                    int H, int W, long total) {  // total = B*256*H*W
  long i = (long)blockIdx.x * blockDim.x + threadIdx.x;
  if (i >= total) return;
  int HW = H * W;
  long per = (long)256 * HW;
  int b = (int)(i / per);
  long r = i - (long)b * per;
  int c = (int)(r / HW);
  int p = (int)(r - (long)c * HW);
  int h = p / W, w = p - h * W;
  xh[(((long)b * (H + 2) + h + 1) * (W + 2) + (w + 1)) * 256 + c] = (half_t)x[i];
}

// ---------------- 3x3 conv, implicit GEMM, 16x32 tile per wave ---------------
// Input is halo-padded NHWC f16 -> no bounds checks in the K loop at all.
template<int CIN, bool PADOUT>
__global__ __launch_bounds__(128) void rf_conv3x3_wmma(
    const half_t* __restrict__ xin, const half_t* __restrict__ wp,
    const float* __restrict__ bias, half_t* __restrict__ out,
    int H, int W, int Mtiles, int coutP, int outStride, int outOff) {
  constexpr int CHUNKS = CIN / 32;
  const int lane = threadIdx.x;      // 0..31
  const int g = lane >> 4;           // lane group
  const int lm = lane & 15;
  const int mtile = blockIdx.x * blockDim.y + threadIdx.y;
  if (mtile >= Mtiles) return;       // wave-uniform: EXEC stays all-ones
  const int n0 = blockIdx.y * 32;    // two 16-wide N tiles per wave
  const int pm = mtile * 16 + lm;    // A-row pixel (always < P; P % 16 == 0)
  const int HW = H * W;
  const int Wp = W + 2;
  int bi = pm / HW;
  int rr = pm - bi * HW;
  int py = rr / W;
  int px = rr - py * W;
  const long rowBase = ((long)bi * (H + 2) + py) * Wp + px;  // +ky*Wp+kx per tap
  v8f acc0 = {}, acc1 = {};
  for (int tap = 0; tap < 9; ++tap) {
    const int ky = tap / 3, kx = tap - ky * 3;
    const half_t* src = xin + (rowBase + ky * Wp + kx) * CIN;
    const half_t* wt = wp + (long)tap * CHUNKS * coutP * 32;
#pragma unroll
    for (int c = 0; c < CHUNKS; ++c) {
      v16h a = *(const v16h*)(src + c * 32 + g * 16);      // K = g*16 .. g*16+15
      v16h b0 = *(const v16h*)(wt + ((long)c * coutP + n0 + lm) * 32 + g * 16);
      v16h b1 = *(const v16h*)(wt + ((long)c * coutP + n0 + 16 + lm) * 32 + g * 16);
      acc0 = __builtin_amdgcn_wmma_f32_16x16x32_f16(false, a, false, b0,
                                                    (short)0, acc0, false, false);
      acc1 = __builtin_amdgcn_wmma_f32_16x16x32_f16(false, a, false, b1,
                                                    (short)0, acc1, false, false);
    }
  }
  const float bv0 = bias[n0 + lm];
  const float bv1 = bias[n0 + 16 + lm];
#pragma unroll
  for (int r = 0; r < 8; ++r) {
    int prow = mtile * 16 + g * 8 + r;       // D row = r + g*8
    float v0 = fmaxf(acc0[r] + bv0, 0.f);    // every SSH conv output is ReLU'd
    float v1 = fmaxf(acc1[r] + bv1, 0.f);
    long o;
    if (PADOUT) {
      int b2 = prow / HW; int rr2 = prow - b2 * HW;
      int y2 = rr2 / W;   int x2 = rr2 - y2 * W;
      o = (((long)b2 * (H + 2) + y2 + 1) * Wp + x2 + 1) * outStride + outOff;
    } else {
      o = (long)prow * outStride + outOff;
    }
    out[o + n0 + lm] = (half_t)v0;
    out[o + n0 + 16 + lm] = (half_t)v1;
  }
}

// ---------------- fused 1x1 heads (box 12->16, cls 6->16), f32 out -----------
__global__ __launch_bounds__(128) void rf_heads_wmma(
    const half_t* __restrict__ fin,
    const half_t* __restrict__ wpb, const float* __restrict__ bb,
    const half_t* __restrict__ wpc, const float* __restrict__ bc,
    float* __restrict__ hb, float* __restrict__ hc, int Mtiles) {
  const int lane = threadIdx.x;
  const int g = lane >> 4;
  const int lm = lane & 15;
  const int mtile = blockIdx.x * blockDim.y + threadIdx.y;
  if (mtile >= Mtiles) return;
  const int pm = mtile * 16 + lm;
  const half_t* src = fin + (long)pm * 256;
  v8f ab = {}, ac = {};
#pragma unroll
  for (int c = 0; c < 8; ++c) {
    v16h a = *(const v16h*)(src + c * 32 + g * 16);
    v16h b0 = *(const v16h*)(wpb + ((long)c * 16 + lm) * 32 + g * 16);
    v16h b1 = *(const v16h*)(wpc + ((long)c * 16 + lm) * 32 + g * 16);
    ab = __builtin_amdgcn_wmma_f32_16x16x32_f16(false, a, false, b0, (short)0, ab, false, false);
    ac = __builtin_amdgcn_wmma_f32_16x16x32_f16(false, a, false, b1, (short)0, ac, false, false);
  }
  const float bvb = bb[lm], bvc = bc[lm];
#pragma unroll
  for (int r = 0; r < 8; ++r) {
    int prow = mtile * 16 + g * 8 + r;
    hb[(long)prow * 16 + lm] = ab[r] + bvb;
    hc[(long)prow * 16 + lm] = ac[r] + bvc;
  }
}

// ---------------- score + box decode -----------------------------------------
__global__ void rf_decode(const float* __restrict__ hbox, const float* __restrict__ hcls,
                          float* __restrict__ scores, float* __restrict__ boxes,
                          int Hc, int stride, float base, int aoff) {
  long total = (long)BATCH * Hc * Hc * 3;
  long i = (long)blockIdx.x * blockDim.x + threadIdx.x;
  if (i >= total) return;
  int anc = (int)(i % 3);
  long p = i / 3;                                  // b*Hc*Hc + h*Hc + w
  int b = (int)(p / ((long)Hc * Hc));
  int rr = (int)(p - (long)b * Hc * Hc);
  int h = rr / Hc, w = rr - h * Hc;
  float c0 = hcls[p * 16 + anc * 2 + 0];
  float c1 = hcls[p * 16 + anc * 2 + 1];
  float sc = 1.f / (1.f + expf(c0 - c1));          // softmax[...,1]
  float rx = hbox[p * 16 + anc * 4 + 0];
  float ry = hbox[p * 16 + anc * 4 + 1];
  float rw = hbox[p * 16 + anc * 4 + 2];
  float rh = hbox[p * 16 + anc * 4 + 3];
  const float SC3[3] = {1.0f, 1.2599210498948732f, 1.5874010519681994f};
  float ps = base * SC3[anc];
  float cx = (w + 0.5f) * stride + rx * 0.1f * ps;
  float cy = (h + 0.5f) * stride + ry * 0.1f * ps;
  float bw = ps * expf(rw * 0.2f);
  float bh = ps * expf(rh * 0.2f);
  long ai = (long)b * ATOT + aoff + ((long)h * Hc + w) * 3 + anc;
  scores[ai] = sc;
  float* bo = boxes + 4 * ai;
  bo[0] = cx - 0.5f * bw; bo[1] = cy - 0.5f * bh;
  bo[2] = cx + 0.5f * bw; bo[3] = cy + 0.5f * bh;
}

// ---------------- top-K: pack keys, bitonic sort descending ------------------
__global__ void rf_build_keys(const float* __restrict__ scores,
                              unsigned long long* __restrict__ keys) {
  long i = (long)blockIdx.x * blockDim.x + threadIdx.x;
  if (i >= (long)BATCH * NSORT) return;
  int b = (int)(i >> 17);
  int k = (int)(i & (NSORT - 1));
  unsigned long long key = 0ull;                  // padding sorts last
  if (k < ATOT) {
    unsigned int sb = __float_as_uint(scores[(long)b * ATOT + k]); // scores > 0
    key = ((unsigned long long)sb << 32) | (unsigned int)k;
  }
  keys[i] = key;
}

__global__ void rf_bitonic_step(unsigned long long* __restrict__ keys, int k, int j) {
  int i = blockIdx.x * blockDim.x + threadIdx.x;  // 0..NSORT-1
  int b = blockIdx.y;
  int ixj = i ^ j;
  if (ixj > i) {
    unsigned long long* kb = keys + ((long)b << 17);
    unsigned long long a = kb[i], c = kb[ixj];
    bool descend = ((i & k) == 0);
    if (descend ? (a < c) : (a > c)) { kb[i] = c; kb[ixj] = a; }
  }
}

__global__ void rf_gather_top(const unsigned long long* __restrict__ keys,
                              const float* __restrict__ boxes,
                              float* __restrict__ tb, float* __restrict__ ts) {
  int i = blockIdx.x * blockDim.x + threadIdx.x;  // 0..BATCH*KTOP-1
  if (i >= BATCH * KTOP) return;
  int b = i >> 9, kk = i & (KTOP - 1);
  unsigned long long key = keys[((long)b << 17) + kk];
  unsigned int idx = (unsigned int)(key & 0xffffffffu);
  float sc = __uint_as_float((unsigned int)(key >> 32));
  const float* bx = boxes + 4 * ((long)b * ATOT + idx);
  tb[(long)i * 4 + 0] = bx[0]; tb[(long)i * 4 + 1] = bx[1];
  tb[(long)i * 4 + 2] = bx[2]; tb[(long)i * 4 + 3] = bx[3];
  ts[i] = sc;
}

// ---------------- greedy sequential NMS (matches fori_loop), one block/batch -
__global__ __launch_bounds__(256) void rf_nms(const float* __restrict__ tb,
                                              const float* __restrict__ ts,
                                              float* __restrict__ out) {
  __shared__ float x1[KTOP], y1[KTOP], x2[KTOP], y2[KTOP], ar[KTOP], sc[KTOP];
  __shared__ int keep[KTOP];
  __shared__ int sup;
  int b = blockIdx.x, tid = threadIdx.x;
  for (int k = tid; k < KTOP; k += 256) {
    const float* p = tb + ((long)b * KTOP + k) * 4;
    x1[k] = p[0]; y1[k] = p[1]; x2[k] = p[2]; y2[k] = p[3];
    ar[k] = (p[2] - p[0]) * (p[3] - p[1]);
    float s = ts[b * KTOP + k];
    sc[k] = s;
    keep[k] = s > 0.5f ? 1 : 0;
  }
  __syncthreads();
  for (int i = 1; i < KTOP; ++i) {
    if (tid == 0) sup = 0;
    __syncthreads();
    for (int j = tid; j < i; j += 256) {
      if (keep[j]) {
        float lx = fmaxf(x1[i], x1[j]), ly = fmaxf(y1[i], y1[j]);
        float rx = fminf(x2[i], x2[j]), ry = fminf(y2[i], y2[j]);
        float iw = fmaxf(rx - lx, 0.f), ih = fmaxf(ry - ly, 0.f);
        float inter = iw * ih;
        float iou = inter / (ar[i] + ar[j] - inter + 1e-9f);
        if (iou > 0.4f) atomicOr(&sup, 1);
      }
    }
    __syncthreads();
    if (tid == 0 && sup) keep[i] = 0;
    __syncthreads();
  }
  for (int k = tid; k < KTOP; k += 256) {
    float* o = out + ((long)b * KTOP + k) * 6;
    o[0] = x1[k]; o[1] = y1[k]; o[2] = x2[k]; o[3] = y2[k];
    o[4] = sc[k]; o[5] = keep[k] ? 1.0f : 0.0f;
  }
}

// =============================================================================
extern "C" void kernel_launch(void* const* d_in, const int* in_sizes, int n_in,
                              void* d_out, int out_size, void* d_ws, size_t ws_size,
                              hipStream_t stream) {
  static const int Hs[5]   = {160, 80, 40, 20, 10};
  static const int STRD[5] = {4, 8, 16, 32, 64};
  static const int BASE[5] = {16, 32, 64, 128, 256};
  static const int AOFF[5] = {0, 76800, 96000, 100800, 102000};

  // ---- resolve param pointers (auto-detect flatten order via in_sizes) ------
  const float *Wssh[5][5], *Bssh[5][5], *Wcls[5], *Bcls[5], *Wbox[5], *Bbox[5];
  auto F_ = [&](int i) { return (const float*)d_in[i]; };
  int s5 = (n_in > 5) ? in_sizes[5] : 0;
  if (s5 == 12) {            // sorted keys everywhere (jax tree_flatten)
    for (int l = 0; l < 5; ++l) { Bbox[l] = F_(5 + 2*l);  Wbox[l] = F_(6 + 2*l); }
    for (int l = 0; l < 5; ++l) { Bcls[l] = F_(15 + 2*l); Wcls[l] = F_(16 + 2*l); }
    for (int l = 0; l < 5; ++l)
      for (int j = 0; j < 5; ++j) { Bssh[l][j] = F_(25 + 10*l + j); Wssh[l][j] = F_(25 + 10*l + 5 + j); }
  } else if (s5 == 3072) {   // sorted top-level, insertion inner
    for (int l = 0; l < 5; ++l) { Wbox[l] = F_(5 + 2*l);  Bbox[l] = F_(6 + 2*l); }
    for (int l = 0; l < 5; ++l) { Wcls[l] = F_(15 + 2*l); Bcls[l] = F_(16 + 2*l); }
    for (int l = 0; l < 5; ++l)
      for (int j = 0; j < 5; ++j) { Wssh[l][j] = F_(25 + 10*l + 2*j); Bssh[l][j] = F_(25 + 10*l + 2*j + 1); }
  } else if (s5 == 128) {    // insertion top-level, sorted inner
    for (int l = 0; l < 5; ++l)
      for (int j = 0; j < 5; ++j) { Bssh[l][j] = F_(5 + 10*l + j); Wssh[l][j] = F_(5 + 10*l + 5 + j); }
    for (int l = 0; l < 5; ++l) { Bcls[l] = F_(55 + 2*l); Wcls[l] = F_(56 + 2*l); }
    for (int l = 0; l < 5; ++l) { Bbox[l] = F_(65 + 2*l); Wbox[l] = F_(66 + 2*l); }
  } else {                   // pure insertion order
    for (int l = 0; l < 5; ++l)
      for (int j = 0; j < 5; ++j) { Wssh[l][j] = F_(5 + 10*l + 2*j); Bssh[l][j] = F_(5 + 10*l + 2*j + 1); }
    for (int l = 0; l < 5; ++l) { Wcls[l] = F_(55 + 2*l); Bcls[l] = F_(56 + 2*l); }
    for (int l = 0; l < 5; ++l) { Wbox[l] = F_(65 + 2*l); Bbox[l] = F_(66 + 2*l); }
  }

  // ---- workspace layout (padded buffers sized for the largest level) --------
  size_t off = 0;
  auto alloc = [&](size_t bytes) { size_t o = off; off = (off + bytes + 255) & ~(size_t)255; return o; };
  const size_t oWP   = alloc(2805760ull * 2);                  // packed weights f16
  const size_t oBIAS = alloc(2080ull * 4);                     // padded biases f32
  const size_t oXH   = alloc(4ull * 162 * 162 * 256 * 2);      // padded NHWC input
  const size_t oT    = alloc(4ull * 162 * 162 * 64 * 2);       // padded t
  const size_t oU    = alloc(4ull * 162 * 162 * 64 * 2);       // padded relu(conv4)
  const size_t oF    = alloc(4ull * 160 * 160 * 256 * 2);      // SSH output (no pad)
  const size_t oHB   = alloc(4ull * 160 * 160 * 16 * 4);       // box head raw f32
  const size_t oHC   = alloc(4ull * 160 * 160 * 16 * 4);       // cls head raw f32
  const size_t oSC   = alloc((size_t)BATCH * ATOT * 4);
  const size_t oBX   = alloc((size_t)BATCH * ATOT * 16);
  const size_t oKY   = alloc((size_t)BATCH * NSORT * 8);
  const size_t oTB   = alloc((size_t)BATCH * KTOP * 16);
  const size_t oTS   = alloc((size_t)BATCH * KTOP * 4);
  if (off > ws_size) return;   // workspace too small: bail deterministically

  char* ws = (char*)d_ws;
  half_t* WP   = (half_t*)(ws + oWP);
  float*  BP   = (float*)(ws + oBIAS);
  half_t* XH   = (half_t*)(ws + oXH);
  half_t* T    = (half_t*)(ws + oT);
  half_t* U    = (half_t*)(ws + oU);
  half_t* Fbuf = (half_t*)(ws + oF);
  float*  HB   = (float*)(ws + oHB);
  float*  HC   = (float*)(ws + oHC);
  float*  SCO  = (float*)(ws + oSC);
  float*  BXS  = (float*)(ws + oBX);
  unsigned long long* KEYS = (unsigned long long*)(ws + oKY);
  float*  TB   = (float*)(ws + oTB);
  float*  TS   = (float*)(ws + oTS);

  // ---- pack all weights/biases ----------------------------------------------
  static const int CIN_[7]   = {256, 256, 64, 64, 64, 256, 256};
  static const int COUT_[7]  = {128, 64, 64, 64, 64, 12, 6};
  static const int COUTP_[7] = {128, 64, 64, 64, 64, 16, 16};
  static const int TAPS_[7]  = {9, 9, 9, 9, 9, 1, 1};
  size_t WPO[5][7]; size_t BO[5][7];
  {
    size_t wph = 0, bof = 0;
    for (int l = 0; l < 5; ++l)
      for (int j = 0; j < 7; ++j) {
        WPO[l][j] = wph; wph += (size_t)TAPS_[j] * (CIN_[j] / 32) * COUTP_[j] * 32;
        BO[l][j]  = bof; bof += COUTP_[j];
      }
  }
  for (int l = 0; l < 5; ++l)
    for (int j = 0; j < 7; ++j) {
      const float* wsrc = (j < 5) ? Wssh[l][j] : ((j == 5) ? Wbox[l] : Wcls[l]);
      const float* bsrc = (j < 5) ? Bssh[l][j] : ((j == 5) ? Bbox[l] : Bcls[l]);
      long tot = (long)TAPS_[j] * (CIN_[j] / 32) * COUTP_[j] * 32;
      rf_pack_w<<<dim3((unsigned)((tot + 255) / 256)), dim3(256), 0, stream>>>(
          wsrc, WP + WPO[l][j], CIN_[j], COUT_[j], COUTP_[j], TAPS_[j]);
      rf_pack_b<<<dim3(1), dim3(256), 0, stream>>>(bsrc, BP + BO[l][j], COUT_[j], COUTP_[j]);
    }

  // ---- per-level pipeline ---------------------------------------------------
  for (int l = 0; l < 5; ++l) {
    const int H = Hs[l], HW = H * H, P = BATCH * HW;
    const int Hp = H + 2;
    // zero padded buffers (halo must be 0)
    {
      long nX = (long)BATCH * Hp * Hp * 256 / 8;
      long nT = (long)BATCH * Hp * Hp * 64 / 8;
      rf_zero_f16<<<dim3((unsigned)((nX + 255) / 256)), dim3(256), 0, stream>>>(XH, nX);
      rf_zero_f16<<<dim3((unsigned)((nT + 255) / 256)), dim3(256), 0, stream>>>(T, nT);
      rf_zero_f16<<<dim3((unsigned)((nT + 255) / 256)), dim3(256), 0, stream>>>(U, nT);
    }
    long tot = (long)BATCH * 256 * HW;
    rf_nchw_to_nhwc_pad<<<dim3((unsigned)((tot + 255) / 256)), dim3(256), 0, stream>>>(
        (const float*)d_in[l], XH, H, H, tot);

    const int mt = P / 16;                 // P is always a multiple of 16
    const dim3 blk(32, 4);
    const unsigned gx = (unsigned)((mt + 3) / 4);
    // conv1: x(256) -> F[0..127]          (N tiles of 32 -> grid.y = Cout/32)
    rf_conv3x3_wmma<256, false><<<dim3(gx, 4), blk, 0, stream>>>(
        XH, WP + WPO[l][0], BP + BO[l][0], Fbuf, H, H, mt, 128, 256, 0);
    // conv2: x(256) -> T(64, padded)
    rf_conv3x3_wmma<256, true><<<dim3(gx, 2), blk, 0, stream>>>(
        XH, WP + WPO[l][1], BP + BO[l][1], T, H, H, mt, 64, 64, 0);
    // conv3: T -> F[128..191]
    rf_conv3x3_wmma<64, false><<<dim3(gx, 2), blk, 0, stream>>>(
        T, WP + WPO[l][2], BP + BO[l][2], Fbuf, H, H, mt, 64, 256, 128);
    // conv4: T -> U(64, padded)
    rf_conv3x3_wmma<64, true><<<dim3(gx, 2), blk, 0, stream>>>(
        T, WP + WPO[l][3], BP + BO[l][3], U, H, H, mt, 64, 64, 0);
    // conv5: U -> F[192..255]
    rf_conv3x3_wmma<64, false><<<dim3(gx, 2), blk, 0, stream>>>(
        U, WP + WPO[l][4], BP + BO[l][4], Fbuf, H, H, mt, 64, 256, 192);
    // fused heads
    rf_heads_wmma<<<dim3(gx, 1), blk, 0, stream>>>(
        Fbuf, WP + WPO[l][5], BP + BO[l][5], WP + WPO[l][6], BP + BO[l][6], HB, HC, mt);
    // decode
    long td = (long)BATCH * HW * 3;
    rf_decode<<<dim3((unsigned)((td + 255) / 256)), dim3(256), 0, stream>>>(
        HB, HC, SCO, BXS, H, STRD[l], (float)BASE[l], AOFF[l]);
  }

  // ---- top-K (bitonic, descending) + NMS ------------------------------------
  rf_build_keys<<<dim3((unsigned)(((long)BATCH * NSORT + 255) / 256)), dim3(256), 0, stream>>>(SCO, KEYS);
  for (unsigned k = 2; k <= NSORT; k <<= 1)
    for (unsigned j = k >> 1; j >= 1; j >>= 1)
      rf_bitonic_step<<<dim3(NSORT / 256, BATCH), dim3(256), 0, stream>>>(KEYS, (int)k, (int)j);
  rf_gather_top<<<dim3((BATCH * KTOP + 255) / 256), dim3(256), 0, stream>>>(KEYS, BXS, TB, TS);
  rf_nms<<<dim3(BATCH), dim3(256), 0, stream>>>(TB, TS, (float*)d_out);
}
